// RechitsGCN_82497731822192
// MI455X (gfx1250) — matrix operand order repaired and checked by reading, not compiled, not run on hardware
//
#include <hip/hip_runtime.h>
#include <hip/hip_bf16.h>

// ---------------------------------------------------------------------------
// RechitsGCN on MI455X (gfx1250): f16 WMMA (16x16x32, f32 accumulate) for all
// 128^3 GEMMs, fp32 VALU for softmax / layernorm / elementwise.
// ---------------------------------------------------------------------------

typedef _Float16 h16;
typedef __attribute__((ext_vector_type(16))) _Float16 v16h;
typedef __attribute__((ext_vector_type(8)))  _Float16 v8h;
typedef __attribute__((ext_vector_type(8)))  float    v8f;

constexpr int G = 1024;   // B*C graphs
constexpr int R = 128;    // rows per graph
constexpr int F = 18;     // input features
constexpr int H = 128;    // hidden

// d_out layout (floats), concatenated in reference return order
constexpr size_t O_OUT   = 0;
constexpr size_t O_SA    = (size_t)G * H;                  // 131072
constexpr size_t O_DENSE = O_SA    + (size_t)G * R * H;
constexpr size_t O_ATTN  = O_DENSE + (size_t)G * R * H;
constexpr size_t O_ADJ   = O_ATTN  + (size_t)G * R * R;

// workspace: 5 f16 slots of G*R*H halves, then normk (G*R f32), then 5 f16 HxH weightsT
constexpr size_t SLOT = (size_t)G * R * H;   // halves

// ---------------------------------------------------------------------------
// WMMA helpers (CDNA5 ISA VGPR layouts, wave32)
// ---------------------------------------------------------------------------
__device__ __forceinline__ v8f wmma_f16(v16h a, v16h b, v8f c) {
    // D = A(16x32 f16) * B(32x16 f16) + C(16x16 f32)
    return __builtin_amdgcn_wmma_f32_16x16x32_f16(
        /*neg_a=*/false, a, /*neg_b=*/false, b,
        /*c_mod=*/(short)0, c, /*reuse_a=*/false, /*reuse_b=*/false);
}

// A fragment: row-major MxK matrix (row stride ld halves).
// lane L: m = m0 + (L&15), g = L>>4.  v16h elem i: k = (i<8 ? g*8+i : 16+g*8+(i-8))
__device__ __forceinline__ v16h load_frag_a(const h16* __restrict__ base,
                                            int m0, int k0, int ld, int lane) {
    int m = m0 + (lane & 15);
    int g = lane >> 4;
    const h16* p = base + (size_t)m * ld + k0 + g * 8;
    v8h lo = *(const v8h*)(p);
    v8h hi = *(const v8h*)(p + 16);
    v16h a;
#pragma unroll
    for (int i = 0; i < 8; ++i) { a[i] = lo[i]; a[i + 8] = hi[i]; }
    return a;
}

// B fragment from B^T stored row-major NxK (row stride ld halves).
// lane L: n = n0 + (L&15), g = L>>4.  v16h elem i: k = (i<8 ? g*8+i : 16+g*8+(i-8))
__device__ __forceinline__ v16h load_frag_bt(const h16* __restrict__ base,
                                             int n0, int k0, int ld, int lane) {
    int n = n0 + (lane & 15);
    int g = lane >> 4;
    const h16* p = base + (size_t)n * ld + k0 + g * 8;
    v8h lo = *(const v8h*)(p);
    v8h hi = *(const v8h*)(p + 16);
    v16h b;
#pragma unroll
    for (int i = 0; i < 8; ++i) { b[i] = lo[i]; b[i + 8] = hi[i]; }
    return b;
}

// row reductions across the 16-lane halves of a wave32 (xor 1,2,4,8 stays in-half)
__device__ __forceinline__ float red_max16(float x) {
    x = fmaxf(x, __shfl_xor(x, 1, 32));
    x = fmaxf(x, __shfl_xor(x, 2, 32));
    x = fmaxf(x, __shfl_xor(x, 4, 32));
    x = fmaxf(x, __shfl_xor(x, 8, 32));
    return x;
}
__device__ __forceinline__ float red_sum16(float x) {
    x += __shfl_xor(x, 1, 32);
    x += __shfl_xor(x, 2, 32);
    x += __shfl_xor(x, 4, 32);
    x += __shfl_xor(x, 8, 32);
    return x;
}

// ---------------------------------------------------------------------------
// Kernel 0: transpose+convert the 5 HxH weights to f16 B^T layout
// ---------------------------------------------------------------------------
__global__ __launch_bounds__(256) void k_prep(const float* __restrict__ wq,
                                              const float* __restrict__ wk,
                                              const float* __restrict__ wv,
                                              const float* __restrict__ w1,
                                              const float* __restrict__ w2,
                                              h16* __restrict__ wT) {
    int idx  = blockIdx.x * 256 + threadIdx.x;       // 5*16384 total
    int wsel = idx >> 14;
    int i    = idx & 16383;
    int n = i >> 7, k = i & 127;
    const float* src = (wsel == 0) ? wq : (wsel == 1) ? wk : (wsel == 2) ? wv
                     : (wsel == 3) ? w1 : w2;
    wT[(size_t)wsel * H * H + n * H + k] = (h16)src[k * H + n];
}

// ---------------------------------------------------------------------------
// Kernel 1: adjacency, degree, norm_k = (deg+1e-6)^-1.5... (norm^3)
// ---------------------------------------------------------------------------
__global__ __launch_bounds__(256) void k_adj(const float* __restrict__ x,
                                             const float* __restrict__ mask,
                                             float* __restrict__ adj_out,
                                             float* __restrict__ normk) {
    int g = blockIdx.x, tid = threadIdx.x;
    __shared__ float cx[R], cy[R], cm[R], degp[256];
    if (tid < R) {
        cx[tid] = x[(size_t)g * R * F + tid * F + 0];
        cy[tid] = x[(size_t)g * R * F + tid * F + 1];
        cm[tid] = mask[(size_t)g * R + tid];
    }
    __syncthreads();
    int r = tid >> 1, s0 = (tid & 1) * 64;
    float xr = cx[r], yr = cy[r], mr = cm[r];
    float acc = 0.f;
    float* arow = adj_out + (size_t)g * R * R + (size_t)r * R;
    for (int t = 0; t < 64; ++t) {
        int s = s0 + t;
        float dx = xr - cx[s], dy = yr - cy[s];
        float d2 = dx * dx + dy * dy;
        d2 = fminf(fmaxf(d2, 1e-12f), 1e12f);
        float a = __expf(-__fsqrt_rn(d2)) * mr * cm[s];
        arow[s] = a;
        acc += a;
    }
    degp[tid] = acc;
    __syncthreads();
    if (tid < R) {
        float deg = degp[2 * tid] + degp[2 * tid + 1];
        float nrm = rsqrtf(deg + 1e-6f);
        normk[(size_t)g * R + tid] = nrm * nrm * nrm;   // K_ITER = 3
    }
}

// ---------------------------------------------------------------------------
// Kernel 2: f_het = x@theta, gate = sigmoid(x@w_t + b_t),
//           fsT[h][s] = f16(f_het[s][h] * norm_k[s])     (K=18: VALU)
// ---------------------------------------------------------------------------
__global__ __launch_bounds__(256) void k_het(const float* __restrict__ x,
                                             const float* __restrict__ theta,
                                             const float* __restrict__ w_t,
                                             const float* __restrict__ b_t,
                                             const float* __restrict__ normk,
                                             h16* __restrict__ fsT,
                                             h16* __restrict__ fhet,
                                             h16* __restrict__ gate) {
    int g = blockIdx.x, tid = threadIdx.x;
    __shared__ float xs[R * F];
    for (int i = tid; i < R * F; i += 256) xs[i] = x[(size_t)g * R * F + i];
    __syncthreads();
    const float* nk = normk + (size_t)g * R;
    for (int t = 0; t < 64; ++t) {
        int idx = t * 256 + tid;
        int r = idx >> 7, h = idx & 127;
        float a0 = 0.f, a1 = 0.f;
#pragma unroll
        for (int f = 0; f < F; ++f) {
            float xv = xs[r * F + f];
            a0 = fmaf(xv, theta[f * H + h], a0);
            a1 = fmaf(xv, w_t[f * H + h], a1);
        }
        float gt = 1.f / (1.f + __expf(-(a1 + b_t[h])));
        fhet[(size_t)g * R * H + idx] = (h16)a0;
        gate[(size_t)g * R * H + idx] = (h16)gt;
        fsT[(size_t)g * R * H + (size_t)h * R + r] = (h16)(a0 * nk[r]);
    }
}

// ---------------------------------------------------------------------------
// Kernel 3: f_hom = relu((adj^3 @ fsT^T) * norm_k[r]);  out_gcn = gate*f_hom+(1-gate)*f_het
// ---------------------------------------------------------------------------
__global__ __launch_bounds__(256) void k_fhom(const float* __restrict__ adj_in,
                                              const float* __restrict__ normk,
                                              const h16* __restrict__ fsT,
                                              const h16* __restrict__ fhet,
                                              const h16* __restrict__ gate,
                                              h16* __restrict__ outg) {
    int g = blockIdx.x, tid = threadIdx.x;
    __shared__ __align__(16) h16 sA[R * R];     // adj^3 as f16 A operand (32 KB)
    __shared__ float snk[R];
    const float* adj = adj_in + (size_t)g * R * R;
    for (int i = tid; i < R * R; i += 256) {
        float a = adj[i];
        sA[i] = (h16)(a * a * a);
    }
    if (tid < R) snk[tid] = normk[(size_t)g * R + tid];
    __syncthreads();

    int wid = tid >> 5, lane = tid & 31;
    int m0 = wid * 16, ml = lane & 15, gl = lane >> 4;
    const h16* Bt = fsT + (size_t)g * R * H;    // [h][s], ld = R
    const h16* fh = fhet + (size_t)g * R * H;
    const h16* gt = gate + (size_t)g * R * H;
    h16* dst = outg + (size_t)g * R * H;

    for (int j = 0; j < 8; ++j) {
        int n0 = j * 16;
        v8f acc = {};
#pragma unroll
        for (int kk = 0; kk < 4; ++kk) {
            v16h a = load_frag_a(sA, m0, kk * 32, R, lane);
            v16h b = load_frag_bt(Bt, n0, kk * 32, R, lane);
            acc = wmma_f16(a, b, acc);
        }
        int n = n0 + ml;
#pragma unroll
        for (int e = 0; e < 8; ++e) {
            int m = m0 + e + 8 * gl;
            float v = fmaxf(acc[e] * snk[m], 0.f);
            float gv = (float)gt[m * H + n];
            float fv = (float)fh[m * H + n];
            dst[m * H + n] = (h16)(gv * v + (1.f - gv) * fv);
        }
    }
}

// ---------------------------------------------------------------------------
// Kernel 4: q/k/v = out_gcn @ {wq,wk,wv}; v stored transposed (vT[h][s])
// ---------------------------------------------------------------------------
__global__ __launch_bounds__(256) void k_qkv(const h16* __restrict__ outg,
                                             const h16* __restrict__ wT,
                                             h16* __restrict__ q,
                                             h16* __restrict__ kmat,
                                             h16* __restrict__ vT) {
    int g = blockIdx.x, which = blockIdx.y, tid = threadIdx.x;
    const h16* A  = outg + (size_t)g * R * H;
    const h16* Bt = wT + (size_t)which * H * H;
    int wid = tid >> 5, lane = tid & 31;
    int m0 = wid * 16, ml = lane & 15, gl = lane >> 4;
    h16* dr = (which == 0) ? q + (size_t)g * R * H
            : (which == 1) ? kmat + (size_t)g * R * H
                           : vT + (size_t)g * R * H;
    for (int j = 0; j < 8; ++j) {
        int n0 = j * 16;
        v8f acc = {};
#pragma unroll
        for (int kk = 0; kk < 4; ++kk) {
            v16h a = load_frag_a(A, m0, kk * 32, H, lane);
            v16h b = load_frag_bt(Bt, n0, kk * 32, H, lane);
            acc = wmma_f16(a, b, acc);
        }
        int n = n0 + ml;
#pragma unroll
        for (int e = 0; e < 8; ++e) {
            int m = m0 + e + 8 * gl;
            if (which == 2) dr[n * R + m] = (h16)acc[e];   // transposed
            else            dr[m * H + n] = (h16)acc[e];
        }
    }
}

// ---------------------------------------------------------------------------
// Kernel 5: logits = q@k^T/sqrt(H) + mask bias; softmax; attn -> f32 out + f16 ws
// ---------------------------------------------------------------------------
__global__ __launch_bounds__(256) void k_attn(const h16* __restrict__ q,
                                              const h16* __restrict__ kmat,
                                              const float* __restrict__ mask,
                                              float* __restrict__ attn_out,
                                              h16* __restrict__ attn16) {
    int g = blockIdx.x, tid = threadIdx.x;
    int wid = tid >> 5, lane = tid & 31;
    int m0 = wid * 16, ml = lane & 15, gl = lane >> 4;
    const h16* A  = q + (size_t)g * R * H;
    const h16* Bt = kmat + (size_t)g * R * H;      // k row-major == (k^T)^T
    v8f acc[8];
#pragma unroll
    for (int j = 0; j < 8; ++j) acc[j] = {};
    for (int kk = 0; kk < 4; ++kk) {
        v16h a = load_frag_a(A, m0, kk * 32, H, lane);
#pragma unroll
        for (int j = 0; j < 8; ++j) {
            v16h b = load_frag_bt(Bt, j * 16, kk * 32, H, lane);
            acc[j] = wmma_f16(a, b, acc[j]);
        }
    }
    const float scale = 0.08838834764831845f;      // 1/sqrt(128)
#pragma unroll
    for (int j = 0; j < 8; ++j) {
        float bias = (1.f - mask[(size_t)g * R + j * 16 + ml]) * -1e9f;
#pragma unroll
        for (int e = 0; e < 8; ++e) acc[j][e] = acc[j][e] * scale + bias;
    }
    float inv[8];
#pragma unroll
    for (int e = 0; e < 8; ++e) {
        float mx = -3.0e38f;
#pragma unroll
        for (int j = 0; j < 8; ++j) mx = fmaxf(mx, acc[j][e]);
        mx = red_max16(mx);
        float sm = 0.f;
#pragma unroll
        for (int j = 0; j < 8; ++j) {
            float p = __expf(acc[j][e] - mx);
            acc[j][e] = p;
            sm += p;
        }
        sm = red_sum16(sm);
        inv[e] = 1.f / sm;
    }
    float* ao = attn_out + (size_t)g * R * R;
    h16*   a16 = attn16 + (size_t)g * R * R;
#pragma unroll
    for (int j = 0; j < 8; ++j) {
        int n = j * 16 + ml;
#pragma unroll
        for (int e = 0; e < 8; ++e) {
            int m = m0 + e + 8 * gl;
            float p = acc[j][e] * inv[e];
            ao[(size_t)m * R + n] = p;
            a16[(size_t)m * R + n] = (h16)p;
        }
    }
}

// ---------------------------------------------------------------------------
// Kernel 6: sa = attn @ v; layernorm(g_sa,be_sa)*mask -> f32 out + f16 ws
// ---------------------------------------------------------------------------
__global__ __launch_bounds__(256) void k_sa(const h16* __restrict__ attn16,
                                            const h16* __restrict__ vT,
                                            const float* __restrict__ mask,
                                            const float* __restrict__ g_sa,
                                            const float* __restrict__ be_sa,
                                            float* __restrict__ sa_out,
                                            h16* __restrict__ sa16) {
    int g = blockIdx.x, tid = threadIdx.x;
    int wid = tid >> 5, lane = tid & 31;
    int m0 = wid * 16, ml = lane & 15, gl = lane >> 4;
    const h16* A  = attn16 + (size_t)g * R * R;    // [r][s], ld R
    const h16* Bt = vT + (size_t)g * R * H;        // [h][s], ld R
    v8f acc[8];
#pragma unroll
    for (int j = 0; j < 8; ++j) acc[j] = {};
    for (int kk = 0; kk < 4; ++kk) {
        v16h a = load_frag_a(A, m0, kk * 32, R, lane);
#pragma unroll
        for (int j = 0; j < 8; ++j) {
            v16h b = load_frag_bt(Bt, j * 16, kk * 32, R, lane);
            acc[j] = wmma_f16(a, b, acc[j]);
        }
    }
    float mean[8], rstd[8];
#pragma unroll
    for (int e = 0; e < 8; ++e) {
        float s = 0.f, qd = 0.f;
#pragma unroll
        for (int j = 0; j < 8; ++j) { float v = acc[j][e]; s += v; qd += v * v; }
        s  = red_sum16(s);
        qd = red_sum16(qd);
        float mu = s * (1.f / H);
        float var = qd * (1.f / H) - mu * mu;
        mean[e] = mu;
        rstd[e] = rsqrtf(var + 1e-3f);
    }
    float* so = sa_out + ((size_t)g * R) * H;
    h16* s16 = sa16 + (size_t)g * R * H;
#pragma unroll
    for (int j = 0; j < 8; ++j) {
        int n = j * 16 + ml;
        float gw = g_sa[n], bw = be_sa[n];
#pragma unroll
        for (int e = 0; e < 8; ++e) {
            int m = m0 + e + 8 * gl;
            float mr = mask[(size_t)g * R + m];
            float v = ((acc[j][e] - mean[e]) * rstd[e] * gw + bw) * mr;
            so[(size_t)m * H + n] = v;
            s16[(size_t)m * H + n] = (h16)v;
        }
    }
}

// ---------------------------------------------------------------------------
// Kernel 7: d = relu(sa @ w1 + b1)
// ---------------------------------------------------------------------------
__global__ __launch_bounds__(256) void k_ffn1(const h16* __restrict__ sa16,
                                              const h16* __restrict__ w1T,
                                              const float* __restrict__ b1,
                                              h16* __restrict__ d16) {
    int g = blockIdx.x, tid = threadIdx.x;
    int wid = tid >> 5, lane = tid & 31;
    int m0 = wid * 16, ml = lane & 15, gl = lane >> 4;
    const h16* A = sa16 + (size_t)g * R * H;
    h16* dst = d16 + (size_t)g * R * H;
    for (int j = 0; j < 8; ++j) {
        int n0 = j * 16;
        v8f acc = {};
#pragma unroll
        for (int kk = 0; kk < 4; ++kk) {
            v16h a = load_frag_a(A, m0, kk * 32, H, lane);
            v16h b = load_frag_bt(w1T, n0, kk * 32, H, lane);
            acc = wmma_f16(a, b, acc);
        }
        int n = n0 + ml;
        float bb = b1[n];
#pragma unroll
        for (int e = 0; e < 8; ++e) {
            int m = m0 + e + 8 * gl;
            dst[m * H + n] = (h16)fmaxf(acc[e] + bb, 0.f);
        }
    }
}

// ---------------------------------------------------------------------------
// Kernel 8: d2 = relu(d @ w2 + b2); dense = LN(d2 + sa)*mask; masked row-mean
// ---------------------------------------------------------------------------
__global__ __launch_bounds__(256) void k_ffn2(const h16* __restrict__ d16,
                                              const h16* __restrict__ w2T,
                                              const float* __restrict__ b2,
                                              const float* __restrict__ sa_f32,
                                              const float* __restrict__ mask,
                                              const float* __restrict__ g_out,
                                              const float* __restrict__ be_out,
                                              float* __restrict__ dense_out,
                                              float* __restrict__ out_mean) {
    int g = blockIdx.x, tid = threadIdx.x;
    int wid = tid >> 5, lane = tid & 31;
    int m0 = wid * 16, ml = lane & 15, gl = lane >> 4;
    __shared__ float scol[H];
    __shared__ float snm;
    if (tid < H) scol[tid] = 0.f;
    if (tid == 0) snm = 0.f;
    __syncthreads();
    if (tid < R) atomicAdd(&snm, mask[(size_t)g * R + tid]);

    const h16* A = d16 + (size_t)g * R * H;
    const float* sres = sa_f32 + (size_t)g * R * H;
    v8f acc[8];
#pragma unroll
    for (int j = 0; j < 8; ++j) acc[j] = {};
    for (int kk = 0; kk < 4; ++kk) {
        v16h a = load_frag_a(A, m0, kk * 32, H, lane);
#pragma unroll
        for (int j = 0; j < 8; ++j) {
            v16h b = load_frag_bt(w2T, j * 16, kk * 32, H, lane);
            acc[j] = wmma_f16(a, b, acc[j]);
        }
    }
    // residual: t = relu(d@w2+b2) + sa
#pragma unroll
    for (int j = 0; j < 8; ++j) {
        int n = j * 16 + ml;
        float bb = b2[n];
#pragma unroll
        for (int e = 0; e < 8; ++e) {
            int m = m0 + e + 8 * gl;
            acc[j][e] = fmaxf(acc[j][e] + bb, 0.f) + sres[(size_t)m * H + n];
        }
    }
    float mean[8], rstd[8];
#pragma unroll
    for (int e = 0; e < 8; ++e) {
        float s = 0.f, qd = 0.f;
#pragma unroll
        for (int j = 0; j < 8; ++j) { float v = acc[j][e]; s += v; qd += v * v; }
        s  = red_sum16(s);
        qd = red_sum16(qd);
        float mu = s * (1.f / H);
        float var = qd * (1.f / H) - mu * mu;
        mean[e] = mu;
        rstd[e] = rsqrtf(var + 1e-3f);
    }
    float* dout = dense_out + (size_t)g * R * H;
#pragma unroll
    for (int j = 0; j < 8; ++j) {
        int n = j * 16 + ml;
        float gw = g_out[n], bw = be_out[n];
        float psum = 0.f;
#pragma unroll
        for (int e = 0; e < 8; ++e) {
            int m = m0 + e + 8 * gl;
            float mr = mask[(size_t)g * R + m];
            float v = ((acc[j][e] - mean[e]) * rstd[e] * gw + bw) * mr;
            dout[(size_t)m * H + n] = v;
            psum += v;
        }
        float tot = psum + __shfl_xor(psum, 16, 32);   // fold the two row-halves
        if (gl == 0) atomicAdd(&scol[n], tot);         // ds_add_f32
    }
    __syncthreads();
    if (tid < H) {
        float nm = snm;
        out_mean[(size_t)g * H + tid] = (nm > 0.f) ? scol[tid] / nm : 0.f;
    }
}

// ---------------------------------------------------------------------------
// Host-side launcher
// ---------------------------------------------------------------------------
extern "C" void kernel_launch(void* const* d_in, const int* in_sizes, int n_in,
                              void* d_out, int out_size, void* d_ws, size_t ws_size,
                              hipStream_t stream) {
    const float* x     = (const float*)d_in[0];
    const float* mask  = (const float*)d_in[1];
    const float* theta = (const float*)d_in[2];
    const float* w_t   = (const float*)d_in[3];
    const float* b_t   = (const float*)d_in[4];
    const float* wq    = (const float*)d_in[5];
    const float* wk    = (const float*)d_in[6];
    const float* wv    = (const float*)d_in[7];
    const float* w1    = (const float*)d_in[8];
    const float* b1    = (const float*)d_in[9];
    const float* w2    = (const float*)d_in[10];
    const float* b2    = (const float*)d_in[11];
    const float* g_sa  = (const float*)d_in[12];
    const float* be_sa = (const float*)d_in[13];
    const float* g_out = (const float*)d_in[14];
    const float* be_out= (const float*)d_in[15];

    float* out = (float*)d_out;

    h16* W  = (h16*)d_ws;
    h16* s0 = W;                 // fsT        -> sa16
    h16* s1 = W + 1 * SLOT;      // out_gcn    -> attn16
    h16* s2 = W + 2 * SLOT;      // f_het16    -> q
    h16* s3 = W + 3 * SLOT;      // gate16     -> k -> d16
    h16* s4 = W + 4 * SLOT;      // vT
    float* nk = (float*)(W + 5 * SLOT);          // G*R floats
    h16* wT   = (h16*)(nk + (size_t)G * R);      // 5 * H*H halves

    k_prep<<<5 * H * H / 256, 256, 0, stream>>>(wq, wk, wv, w1, w2, wT);
    k_adj <<<G, 256, 0, stream>>>(x, mask, out + O_ADJ, nk);
    k_het <<<G, 256, 0, stream>>>(x, theta, w_t, b_t, nk, s0, s2, s3);
    k_fhom<<<G, 256, 0, stream>>>(out + O_ADJ, nk, s0, s2, s3, s1);
    k_qkv <<<dim3(G, 3), 256, 0, stream>>>(s1, wT, s2, s3, s4);
    k_attn<<<G, 256, 0, stream>>>(s2, s3, mask, out + O_ATTN, s1);
    k_sa  <<<G, 256, 0, stream>>>(s1, s4, mask, g_sa, be_sa, out + O_SA, s0);
    k_ffn1<<<G, 256, 0, stream>>>(s0, wT + 3 * (size_t)H * H, b1, s3);
    k_ffn2<<<G, 256, 0, stream>>>(s3, wT + 4 * (size_t)H * H, b2,
                                  out + O_SA, mask, g_out, be_out,
                                  out + O_DENSE, out + O_OUT);
    (void)in_sizes; (void)n_in; (void)out_size; (void)ws_size;
}
